// TransformerLayer_88948772700541
// MI455X (gfx1250) — matrix operand order repaired and checked by reading, not compiled, not run on hardware
//
#include <hip/hip_runtime.h>
#include <math.h>

typedef _Float16 v16h __attribute__((ext_vector_type(16)));
typedef _Float16 v8h  __attribute__((ext_vector_type(8)));
typedef float    v8f  __attribute__((ext_vector_type(8)));

namespace {

constexpr int Bc = 4, Nn = 128, NDc = 256, EDc = 128, CDc = 256, Hc = 8;

__device__ __forceinline__ float lrelu(float x){ return x > 0.f ? x : 0.1f * x; }

// A-fragment: 16x32 f16 tile from row-major f16 source (lda in halves).
// Lane layout per CDNA5 ISA 7.12.2: lane m=l&15, halves j<8 -> K = k0+8*(l>>4)+j,
// halves j>=8 -> K = k0+16+8*(l>>4)+(j-8).
__device__ __forceinline__ v16h load_a_frag(const _Float16* A, int lda, int row0, int k0, int lane){
  const int m  = lane & 15;
  const int hi = lane >> 4;
  const _Float16* base = A + (size_t)(row0 + m) * lda + k0 + 8 * hi;
  v8h x0 = *(const v8h*)(base);
  v8h x1 = *(const v8h*)(base + 16);
  v16h r;
#pragma unroll
  for (int j = 0; j < 8; ++j){ r[j] = x0[j]; r[8 + j] = x1[j]; }
  return r;
}

// B-fragment: 32x16 f16 tile read from TRANSPOSED weights WT[ncols][K] (ldb = K).
// Lane holds column n=l&15, halves j<8 -> K = k0+8*(l>>4)+j, j>=8 -> K = k0+16+8*(l>>4)+(j-8).
__device__ __forceinline__ v16h load_b_frag(const _Float16* BT, int ldb, int col0, int k0, int lane){
  const int n  = lane & 15;
  const int hi = lane >> 4;
  const _Float16* base = BT + (size_t)(col0 + n) * ldb + k0 + 8 * hi;
  v8h x0 = *(const v8h*)(base);
  v8h x1 = *(const v8h*)(base + 16);
  v16h r;
#pragma unroll
  for (int j = 0; j < 8; ++j){ r[j] = x0[j]; r[8 + j] = x1[j]; }
  return r;
}

__device__ __forceinline__ v8f wmma16(v16h a, v16h b, v8f c){
  return __builtin_amdgcn_wmma_f32_16x16x32_f16(false, a, false, b, (short)0, c, false, false);
}

// ---------------- prep: transpose+convert fp32 weights W[K][Ncols] -> f16 WT[Ncols][K]
__global__ void k_wt(const float* __restrict__ W, _Float16* __restrict__ WT, int K, int Ncols){
  int i = blockIdx.x * blockDim.x + threadIdx.x;
  if (i >= K * Ncols) return;
  int k = i / Ncols, n = i - k * Ncols;
  WT[(size_t)n * K + k] = (_Float16)W[i];
}

// ---------------- FiLM: film[b][0:1024] = conds[b] @ Wc + bc
__global__ void k_film(const float* __restrict__ conds, const float* __restrict__ Wc,
                       const float* __restrict__ bc, float* __restrict__ film){
  int idx = blockIdx.x * blockDim.x + threadIdx.x;
  if (idx >= Bc * 1024) return;
  int b = idx >> 10, c = idx & 1023;
  const float* cd = conds + b * CDc;
  float acc = bc[c];
  for (int k = 0; k < CDc; ++k) acc += cd[k] * Wc[k * 1024 + c];
  film[idx] = acc;
}

// ---------------- n_in = lrelu(LN(nodes)*(1+mul_in)+add_in)  (stored f16)
__global__ void k_nin(const float* __restrict__ nodes, const float* __restrict__ film,
                      _Float16* __restrict__ nin){
  __shared__ float rA[256], rB[256];
  const int row = blockIdx.x;
  const int b = row >> 7;
  const int c = threadIdx.x;
  float x = nodes[(size_t)row * NDc + c];
  rA[c] = x; rB[c] = x * x;
  __syncthreads();
  for (int s = 128; s > 0; s >>= 1){
    if (c < s){ rA[c] += rA[c + s]; rB[c] += rB[c + s]; }
    __syncthreads();
  }
  const float mean = rA[0] * (1.f / NDc);
  const float var  = rB[0] * (1.f / NDc) - mean * mean;
  float y = (x - mean) * rsqrtf(var + 1e-5f);
  const float m = film[b * 1024 + c];
  const float a = film[b * 1024 + 256 + c];
  nin[(size_t)row * NDc + c] = (_Float16)lrelu(y * (1.f + m) + a);
}

// ---------------- qkv = n_in @ Wqkv + bqkv   (512 x 768, K=256) via WMMA
__global__ __launch_bounds__(128) void k_qkv(const _Float16* __restrict__ nin,
                                             const _Float16* __restrict__ WqkvT,
                                             const float* __restrict__ bqkv,
                                             float* __restrict__ qkv){
  const int row0 = blockIdx.x * 16;           // 32 row tiles
  const int cg   = blockIdx.y;                // 6 groups of 128 cols
  const int lane = threadIdx.x & 31, wave = threadIdx.x >> 5;
  for (int s = 0; s < 2; ++s){
    const int col0 = cg * 128 + (wave * 2 + s) * 16;
    v8f c = {};
    for (int k0 = 0; k0 < 256; k0 += 32){
      v16h a = load_a_frag(nin, NDc, row0, k0, lane);
      v16h b = load_b_frag(WqkvT, NDc, col0, k0, lane);
      c = wmma16(a, b, c);
    }
    const int n = lane & 15, mh = lane >> 4;
    const int col = col0 + n;
    const float bias = bqkv[col];
#pragma unroll
    for (int r = 0; r < 8; ++r){
      const int row = row0 + r + 8 * mh;
      qkv[(size_t)row * 768 + col] = c[r] + bias;
    }
  }
}

// ---------------- fully fused edge path (one global read + one global write of edges):
// e_in = lrelu(LN(edges)); ss = e_in@Wss+bss; new_e = q*k*(1+scale)+shift;
// logits = sum_d(new_e)/sqrt(D); edges1 = edges + lrelu(new_e)@Weo + beo;
// out_edges = edges1 + lrelu(lrelu(LN(edges1))@We1+be1)@We2 + be2
__global__ __launch_bounds__(128) void k_edge_core(
    const float* __restrict__ edges, const float* __restrict__ qkv,
    const _Float16* __restrict__ WssT, const float* __restrict__ bss,
    const _Float16* __restrict__ WeoT, const float* __restrict__ beo,
    const _Float16* __restrict__ We1T, const float* __restrict__ be1,
    const _Float16* __restrict__ We2T, const float* __restrict__ be2,
    float* __restrict__ logits, float* __restrict__ out_edges){
  __shared__ alignas(16) _Float16 A1[16 * 128];   // f16 LN tile (reused twice)
  __shared__ float NE[16 * 256];                  // new_e tile (f32)
  __shared__ alignas(16) _Float16 A2[16 * 256];   // f16 activation tile (reused)
  __shared__ float E[16 * 128];                   // raw edges tile -> edges1 tile
  __shared__ float rA[128], rB[128];

  const int bid  = blockIdx.x;      // 4096
  const int b    = bid >> 10;
  const int tile = bid & 1023;
  const int p0   = tile * 16;       // 16 consecutive (q,k) pairs -> same q
  const int qi   = p0 >> 7;
  const int kk0  = p0 & 127;
  const int tid  = threadIdx.x;
  const int lane = tid & 31, wave = tid >> 5;
  const int r8 = tid >> 3, s8 = tid & 7;          // 8 threads per row for row ops

  // stage 1: load edges rows -> E (raw f32) and A1 = f16 lrelu(LN(row))
  {
    const float* erow = edges + ((((size_t)b * Nn + qi) * Nn) + kk0 + r8) * EDc;
    float vals[16]; float sum = 0.f, sq = 0.f;
#pragma unroll
    for (int j = 0; j < 16; ++j){
      float x = erow[s8 * 16 + j];
      vals[j] = x; E[r8 * 128 + s8 * 16 + j] = x; sum += x; sq += x * x;
    }
    rA[tid] = sum; rB[tid] = sq;
    __syncthreads();
    if (s8 == 0){
      float S = 0.f, SQ = 0.f;
      for (int j = 0; j < 8; ++j){ S += rA[r8 * 8 + j]; SQ += rB[r8 * 8 + j]; }
      rA[r8 * 8] = S; rB[r8 * 8] = SQ;
    }
    __syncthreads();
    const float mean = rA[r8 * 8] * (1.f / EDc);
    const float var  = rB[r8 * 8] * (1.f / EDc) - mean * mean;
    const float inv  = rsqrtf(var + 1e-5f);
#pragma unroll
    for (int j = 0; j < 16; ++j)
      A1[r8 * 128 + s8 * 16 + j] = (_Float16)lrelu((vals[j] - mean) * inv);
  }
  __syncthreads();

  // stage 2: ss GEMM (K=128) in paired shift/scale col-tiles, combine with q*k
  for (int pt = 0; pt < 4; ++pt){
    const int ts = wave * 4 + pt;     // shift tile 0..15 ; scale tile = ts+16
    v8f cs = {}; v8f cc = {};
    for (int k0 = 0; k0 < 128; k0 += 32){
      v16h a  = load_a_frag(A1, 128, 0, k0, lane);
      v16h b0 = load_b_frag(WssT, 128, ts * 16, k0, lane);
      v16h b1 = load_b_frag(WssT, 128, 256 + ts * 16, k0, lane);
      cs = wmma16(a, b0, cs);
      cc = wmma16(a, b1, cc);
    }
    const int n = lane & 15, mh = lane >> 4;
    const int c = ts * 16 + n;                       // channel 0..255 (= h*32+d)
    const float qv  = qkv[((size_t)(b * Nn + qi)) * 768 + c];
    const float bsh = bss[c], bsc = bss[256 + c];
#pragma unroll
    for (int r = 0; r < 8; ++r){
      const int pl = r + 8 * mh;
      const int kk = kk0 + pl;
      const float kv = qkv[((size_t)(b * Nn + kk)) * 768 + 256 + c];
      const float qk = qv * kv;
      NE[pl * 256 + c] = qk * (1.f + (cc[r] + bsc)) + (cs[r] + bsh);
    }
  }
  __syncthreads();

  // stage 3: logits[b][h][q][k] = sum_d NE / sqrt(32)
  {
    float s = 0.f;
#pragma unroll
    for (int d = 0; d < 32; ++d) s += NE[r8 * 256 + (tid & 7) * 32 + d];
    logits[(((size_t)b * Hc + (tid & 7)) * Nn + qi) * Nn + kk0 + r8] = s * 0.17677669529663687f;
  }

  // stage 4: A2 = lrelu(NE) as f16
  for (int i = tid; i < 16 * 256; i += 128) A2[i] = (_Float16)lrelu(NE[i]);
  __syncthreads();

  // stage 5: edges1 = edges + A2 @ WeoT + beo  (K=256, 128 cols), in-place into E
  for (int s2 = 0; s2 < 2; ++s2){
    const int col0 = (wave * 2 + s2) * 16;
    v8f c2 = {};
    for (int k0 = 0; k0 < 256; k0 += 32){
      v16h a  = load_a_frag(A2, 256, 0, k0, lane);
      v16h bb = load_b_frag(WeoT, 256, col0, k0, lane);
      c2 = wmma16(a, bb, c2);
    }
    const int n = lane & 15, mh = lane >> 4;
    const float bias = beo[col0 + n];
#pragma unroll
    for (int r = 0; r < 8; ++r){
      const int pl = r + 8 * mh;
      const int idx = pl * 128 + col0 + n;
      E[idx] = E[idx] + c2[r] + bias;     // each idx owned by exactly one thread
    }
  }
  __syncthreads();

  // stage 6: A1 = f16 lrelu(LN(edges1 rows))
  {
    float vals[16]; float sum = 0.f, sq = 0.f;
#pragma unroll
    for (int j = 0; j < 16; ++j){
      float x = E[r8 * 128 + s8 * 16 + j];
      vals[j] = x; sum += x; sq += x * x;
    }
    rA[tid] = sum; rB[tid] = sq;
    __syncthreads();
    if (s8 == 0){
      float S = 0.f, SQ = 0.f;
      for (int j = 0; j < 8; ++j){ S += rA[r8 * 8 + j]; SQ += rB[r8 * 8 + j]; }
      rA[r8 * 8] = S; rB[r8 * 8] = SQ;
    }
    __syncthreads();
    const float mean = rA[r8 * 8] * (1.f / EDc);
    const float var  = rB[r8 * 8] * (1.f / EDc) - mean * mean;
    const float inv  = rsqrtf(var + 1e-5f);
#pragma unroll
    for (int j = 0; j < 16; ++j)
      A1[r8 * 128 + s8 * 16 + j] = (_Float16)lrelu((vals[j] - mean) * inv);
  }
  __syncthreads();

  // stage 7: A2[0:16x128] = f16 lrelu(A1 @ We1T + be1)   (K=128)
  for (int s2 = 0; s2 < 2; ++s2){
    const int col0 = (wave * 2 + s2) * 16;
    v8f c = {};
    for (int k0 = 0; k0 < 128; k0 += 32)
      c = wmma16(load_a_frag(A1, 128, 0, k0, lane), load_b_frag(We1T, 128, col0, k0, lane), c);
    const int n = lane & 15, mh = lane >> 4;
    const float bias = be1[col0 + n];
#pragma unroll
    for (int r = 0; r < 8; ++r){
      const int pl = r + 8 * mh;
      A2[pl * 128 + col0 + n] = (_Float16)lrelu(c[r] + bias);
    }
  }
  __syncthreads();

  // stage 8: out_edges = edges1 + A2 @ We2T + be2   (K=128)
  for (int s2 = 0; s2 < 2; ++s2){
    const int col0 = (wave * 2 + s2) * 16;
    v8f c = {};
    for (int k0 = 0; k0 < 128; k0 += 32)
      c = wmma16(load_a_frag(A2, 128, 0, k0, lane), load_b_frag(We2T, 128, col0, k0, lane), c);
    const int n = lane & 15, mh = lane >> 4;
    const int col = col0 + n;
    const float bias = be2[col];
#pragma unroll
    for (int r = 0; r < 8; ++r){
      const int pl = r + 8 * mh;
      const size_t off = ((((size_t)b * Nn + qi) * Nn) + kk0 + pl) * EDc + col;
      out_edges[off] = E[pl * 128 + col] + c[r] + bias;
    }
  }
}

// ---------------- softmax over q axis (axis=2 of (b,h,q,k))
__global__ void k_softmax(const float* __restrict__ logits, float* __restrict__ attn){
  const int bh = blockIdx.x;              // 32
  const int k  = threadIdx.x;             // 128
  const float* L = logits + (size_t)bh * Nn * Nn;
  float* A = attn + (size_t)bh * Nn * Nn;
  float mx = -1e30f;
  for (int q = 0; q < Nn; ++q) mx = fmaxf(mx, L[q * Nn + k]);
  float s = 0.f;
  for (int q = 0; q < Nn; ++q) s += expf(L[q * Nn + k] - mx);
  const float inv = 1.f / s;
  for (int q = 0; q < Nn; ++q) A[q * Nn + k] = expf(L[q * Nn + k] - mx) * inv;
}

// ---------------- wv[b,q,h*32+d] = sum_k attn[b,h,q,k] * v[b,k,h*32+d]
__global__ void k_wv(const float* __restrict__ attn, const float* __restrict__ qkv,
                     float* __restrict__ wv){
  __shared__ float at[Hc * Nn];
  const int row = blockIdx.x;            // b*N+q
  const int b = row >> 7, q = row & 127;
  const int c = threadIdx.x;             // 256
  const int h = c >> 5;
  for (int i = c; i < Hc * Nn; i += 256){
    int hh = i >> 7, k = i & 127;
    at[i] = attn[(((size_t)b * Hc + hh) * Nn + q) * Nn + k];
  }
  __syncthreads();
  float acc = 0.f;
  for (int k = 0; k < Nn; ++k)
    acc += at[h * Nn + k] * qkv[((size_t)(b * Nn + k)) * 768 + 512 + c];
  wv[(size_t)row * 256 + c] = acc;
}

// ---------------- node path: Wno proj + residual, FiLM-LN, MLP, residual
__global__ void k_nodes(const float* __restrict__ nodes, const float* __restrict__ wv,
                        const float* __restrict__ film,
                        const float* __restrict__ Wno, const float* __restrict__ bno,
                        const float* __restrict__ Wn1, const float* __restrict__ bn1,
                        const float* __restrict__ Wn2, const float* __restrict__ bn2,
                        float* __restrict__ out_nodes){
  __shared__ float wvr[256], nor[256], h1[1024], rA[256], rB[256];
  const int row = blockIdx.x, b = row >> 7, c = threadIdx.x;
  wvr[c] = wv[(size_t)row * 256 + c];
  __syncthreads();
  float acc = bno[c];
  for (int k = 0; k < 256; ++k) acc += wvr[k] * Wno[k * 256 + c];
  const float nd = nodes[(size_t)row * 256 + c] + acc;   // nodes after attn residual
  rA[c] = nd; rB[c] = nd * nd;
  __syncthreads();
  for (int s = 128; s > 0; s >>= 1){
    if (c < s){ rA[c] += rA[c + s]; rB[c] += rB[c + s]; }
    __syncthreads();
  }
  const float mean = rA[0] * (1.f / 256), var = rB[0] * (1.f / 256) - mean * mean;
  float y = (nd - mean) * rsqrtf(var + 1e-5f);
  y = lrelu(y * (1.f + film[b * 1024 + 512 + c]) + film[b * 1024 + 768 + c]);
  nor[c] = y;
  __syncthreads();
  for (int j = 0; j < 4; ++j){
    const int col = j * 256 + c;
    float a2 = bn1[col];
    for (int k = 0; k < 256; ++k) a2 += nor[k] * Wn1[k * 1024 + col];
    h1[col] = lrelu(a2);
  }
  __syncthreads();
  float a3 = bn2[c];
  for (int k = 0; k < 1024; ++k) a3 += h1[k] * Wn2[k * 256 + c];
  out_nodes[(size_t)row * 256 + c] = nd + a3;
}

} // namespace

extern "C" void kernel_launch(void* const* d_in, const int* in_sizes, int n_in,
                              void* d_out, int out_size, void* d_ws, size_t ws_size,
                              hipStream_t stream) {
  (void)in_sizes; (void)n_in; (void)out_size; (void)ws_size;
  const float* nodes = (const float*)d_in[0];
  const float* edges = (const float*)d_in[1];
  const float* conds = (const float*)d_in[2];
  const float* Wc   = (const float*)d_in[3];  const float* bc   = (const float*)d_in[4];
  const float* Wqkv = (const float*)d_in[5];  const float* bqkv = (const float*)d_in[6];
  const float* Wss  = (const float*)d_in[7];  const float* bss  = (const float*)d_in[8];
  const float* Wno  = (const float*)d_in[9];  const float* bno  = (const float*)d_in[10];
  const float* Weo  = (const float*)d_in[11]; const float* beo  = (const float*)d_in[12];
  const float* Wn1  = (const float*)d_in[13]; const float* bn1  = (const float*)d_in[14];
  const float* Wn2  = (const float*)d_in[15]; const float* bn2  = (const float*)d_in[16];
  const float* We1  = (const float*)d_in[17]; const float* be1  = (const float*)d_in[18];
  const float* We2  = (const float*)d_in[19]; const float* be2  = (const float*)d_in[20];

  char* ws = (char*)d_ws;                       // ~7.25 MB total
  float*    film   = (float*)   (ws + 0);        // 4*1024 f32
  float*    qkv    = (float*)   (ws + 16384);    // 512*768 f32
  float*    logits = (float*)   (ws + 1589248);  // 4*8*128*128 f32
  float*    attn   = (float*)   (ws + 3686400);  // 4*8*128*128 f32
  float*    wv     = (float*)   (ws + 5783552);  // 512*256 f32
  _Float16* nin16  = (_Float16*)(ws + 6307840);  // 512*256 f16
  _Float16* WssT   = (_Float16*)(ws + 6569984);  // 512x128 f16
  _Float16* WeoT   = (_Float16*)(ws + 6701056);  // 128x256 f16
  _Float16* We1T   = (_Float16*)(ws + 6766592);  // 128x128 f16
  _Float16* We2T   = (_Float16*)(ws + 6799360);  // 128x128 f16
  _Float16* WqkvT  = (_Float16*)(ws + 6831104);  // 768x256 f16

  float* out_nodes = (float*)d_out;
  float* out_edges = out_nodes + Bc * Nn * NDc;  // 131072 offset

  k_wt<<<256, 256, 0, stream>>>(Wss,  WssT,  EDc, 2 * NDc);
  k_wt<<<128, 256, 0, stream>>>(Weo,  WeoT,  NDc, EDc);
  k_wt<<<64,  256, 0, stream>>>(We1,  We1T,  EDc, EDc);
  k_wt<<<64,  256, 0, stream>>>(We2,  We2T,  EDc, EDc);
  k_wt<<<768, 256, 0, stream>>>(Wqkv, WqkvT, NDc, 3 * NDc);

  k_film<<<16, 256, 0, stream>>>(conds, Wc, bc, film);
  k_nin<<<Bc * Nn, 256, 0, stream>>>(nodes, film, nin16);
  k_qkv<<<dim3(32, 6), 128, 0, stream>>>(nin16, WqkvT, bqkv, qkv);
  k_edge_core<<<4096, 128, 0, stream>>>(edges, qkv, WssT, bss, WeoT, beo,
                                        We1T, be1, We2T, be2, logits, out_edges);
  k_softmax<<<Bc * Hc, 128, 0, stream>>>(logits, attn);
  k_wv<<<Bc * Nn, 256, 0, stream>>>(attn, qkv, wv);
  k_nodes<<<Bc * Nn, 256, 0, stream>>>(nodes, wv, film, Wno, bno, Wn1, bn1, Wn2, bn2, out_nodes);
}
